// HierarchyGCN_6743098655307
// MI455X (gfx1250) — matrix operand order repaired and checked by reading, not compiled, not run on hardware
//
#include <hip/hip_runtime.h>
#include <math.h>

typedef __attribute__((ext_vector_type(16))) _Float16 v16h;
typedef __attribute__((ext_vector_type(8)))  float    v8f;

#define B_    256
#define L_    256
#define D_    512
#define LT    64     // l-rows per workgroup
#define TDG   256    // d-cols per workgroup
#define NTHR  512    // 16 waves (wave32): 4 l-slices x 4 d-groups
#define NWAVE 16
#define DTW   4      // 16-wide d-tiles per wave (4*16 = 64 cols per wave)

// A-matrix (16-bit, 16x32 MxK) K index for v16h element e, lane half h.
// ISA 7.12.2: v0: K=0,1|8,9  v1: K=2,3|10,11  v2: K=4,5|12,13  v3: K=6,7|14,15
//             v4..7: K=16..23 | 24..31
__device__ __forceinline__ int kmapA(int e, int h) {
    const int vg = e >> 1;
    return ((vg < 4) ? 0 : 16) + ((vg & 3) << 1) + (e & 1) + (h << 3);
}
// B-matrix (16-bit, 32x16 KxN): same K order as A: lanes 0-15 hold K=0..7|16..23,
// lanes 16-31 hold K=8..15|24..31.
__device__ __forceinline__ int kmapB(int e, int h) {
    return ((e < 8) ? e : e + 8) + (h << 3);
}

__global__ __launch_bounds__(NTHR)
void hgcn_kernel(const float* __restrict__ X,
                 const float* __restrict__ Ptd,
                 const float* __restrict__ Pbu,
                 const float* __restrict__ td_b1,
                 const float* __restrict__ td_b2,
                 const float* __restrict__ bu_b1,
                 const float* __restrict__ bu_b2,
                 const float* __restrict__ td_w,
                 const float* __restrict__ bu_w,
                 const float* __restrict__ loop_w,
                 float* __restrict__ out)
{
    __shared__ float sTd[LT], sBu[LT], sLp[LT];

    const int dbase    = blockIdx.x * TDG;    // d-column group of this workgroup
    const int l0       = blockIdx.y * LT;     // l-row tile of this workgroup
    const int b        = blockIdx.z;
    const int tid      = threadIdx.x;
    const int wave     = tid >> 5;            // 0..15
    const int ls       = wave >> 2;           // l-slice 0..3  (16 rows each)
    const int dg       = wave & 3;            // d-group 0..3  (64 cols each)
    const int lane     = tid & 31;
    const int half_sel = lane >> 4;
    const int lane16   = lane & 15;

    const float* __restrict__ Xb = X + (long)b * L_ * D_;

    // ---------- Phase 1: gate dot-products + sigmoid (full f32, full D) ----------
    for (int r = wave; r < LT; r += NWAVE) {
        const int l = l0 + r;
        const float* __restrict__ row = Xb + (long)l * D_;
        const float b2td = td_b2[l];
        const float b2bu = bu_b2[l];
        float atd = 0.f, abu = 0.f, alp = 0.f;
        for (int d = lane; d < D_; d += 32) {
            const float x = row[d];
            atd += (x + b2td) * td_w[d];
            abu += (x + b2bu) * bu_w[d];
            alp += x * loop_w[d];
        }
        #pragma unroll
        for (int off = 16; off > 0; off >>= 1) {
            atd += __shfl_xor(atd, off, 32);
            abu += __shfl_xor(abu, off, 32);
            alp += __shfl_xor(alp, off, 32);
        }
        if (lane == 0) {
            sTd[r] = 1.f / (1.f + __expf(-atd));
            sBu[r] = 1.f / (1.f + __expf(-abu));
            sLp[r] = 1.f / (1.f + __expf(-alp));
        }
    }
    __syncthreads();

    // ---------- Phase 2: two GEMMs via v_wmma_f32_16x16x32_f16 ----------
    v8f acc_td[DTW], acc_bu[DTW];
    #pragma unroll
    for (int j = 0; j < DTW; ++j) { v8f z = {}; acc_td[j] = z; acc_bu[j] = z; }

    const long aRowBase = (long)(l0 + ls * 16 + lane16) * L_;

    for (int m0 = 0; m0 < L_; m0 += 32) {
        v16h aTd, aBu;
        #pragma unroll
        for (int e = 0; e < 16; ++e) {
            const int k = kmapA(e, half_sel);
            aTd[e] = (_Float16)Ptd[aRowBase + m0 + k];
            aBu[e] = (_Float16)Pbu[aRowBase + m0 + k];
        }
        #pragma unroll
        for (int j = 0; j < DTW; ++j) {
            const int col = dbase + ((dg * DTW + j) << 4) + lane16;
            v16h bf;
            #pragma unroll
            for (int e = 0; e < 16; ++e) {
                const int k = kmapB(e, half_sel);
                bf[e] = (_Float16)Xb[(long)(m0 + k) * D_ + col];
            }
            acc_td[j] = __builtin_amdgcn_wmma_f32_16x16x32_f16(
                false, aTd, false, bf, (short)0, acc_td[j], false, false);
            acc_bu[j] = __builtin_amdgcn_wmma_f32_16x16x32_f16(
                false, aBu, false, bf, (short)0, acc_bu[j], false, false);
        }
    }

    // ---------- Phase 3: bias + sigmoid gating + loop msg + ReLU (f32) ----------
    const long outBase = (long)b * L_ * D_;
    #pragma unroll
    for (int j = 0; j < DTW; ++j) {
        const int col = dbase + ((dg * DTW + j) << 4) + lane16;
        #pragma unroll
        for (int v = 0; v < 8; ++v) {
            const int rr = ls * 16 + v + (half_sel << 3);   // row within l-tile
            const int l  = l0 + rr;
            const long ld = (long)l * D_ + col;
            const float td = (acc_td[j][v] + td_b1[ld]) * sTd[rr];
            const float bu = (acc_bu[j][v] + bu_b1[ld]) * sBu[rr];
            const float lp = Xb[ld] * sLp[rr];
            const float r  = td + bu + lp;
            out[outBase + ld] = r > 0.f ? r : 0.f;
        }
    }
}

extern "C" void kernel_launch(void* const* d_in, const int* in_sizes, int n_in,
                              void* d_out, int out_size, void* d_ws, size_t ws_size,
                              hipStream_t stream) {
    (void)in_sizes; (void)n_in; (void)out_size; (void)d_ws; (void)ws_size;
    const float* X      = (const float*)d_in[0];
    const float* Ptd    = (const float*)d_in[1];
    const float* Pbu    = (const float*)d_in[2];
    const float* td_b1  = (const float*)d_in[3];
    const float* td_b2  = (const float*)d_in[4];
    const float* bu_b1  = (const float*)d_in[5];
    const float* bu_b2  = (const float*)d_in[6];
    const float* td_w   = (const float*)d_in[7];
    const float* bu_w   = (const float*)d_in[8];
    const float* loop_w = (const float*)d_in[9];
    float* outp = (float*)d_out;

    // x,y fast-varying so the 8 workgroups sharing batch b are adjacent -> X_b L2 reuse
    dim3 grid(D_ / TDG, L_ / LT, B_);
    dim3 block(NTHR);
    hgcn_kernel<<<grid, block, 0, stream>>>(X, Ptd, Pbu, td_b1, td_b2,
                                            bu_b1, bu_b2, td_w, bu_w, loop_w, outp);
}
